// MHA_36584531427723
// MI455X (gfx1250) — matrix operand order repaired and checked by reading, not compiled, not run on hardware
//
#include <hip/hip_runtime.h>
#include <hip/hip_bf16.h>

typedef __attribute__((ext_vector_type(16))) _Float16 v16h;
typedef __attribute__((ext_vector_type(8)))  float    v8f;

#define DIMV 1024
#define HEADS 16
#define HD 64
#define BATCH 2
#define NQ 2048
#define NK 2048

// ------------------------------------------------ CDNA5 async global->LDS copy
// GLOBAL_LOAD_ASYNC_TO_LDS_B128 (GV mode): vdst = LDS byte offset (low 32 bits
// of a generic shared pointer), vaddr = 64-bit global address. Tracked by
// ASYNCcnt; completed with s_wait_asynccnt before the workgroup barrier.
__device__ __forceinline__ void async_copy_b128(const _Float16* gsrc, _Float16* ldst) {
    unsigned int loff = (unsigned int)(uintptr_t)ldst;
    asm volatile("global_load_async_to_lds_b128 %0, %1, off"
                 :: "v"(loff), "v"(gsrc) : "memory");
}
__device__ __forceinline__ void wait_async() {
    asm volatile("s_wait_asynccnt 0" ::: "memory");
}

// ---------------------------------------------------------------- f32 -> f16
__global__ void cvt_f32_f16(const float* __restrict__ x, _Float16* __restrict__ y, int n) {
    int i = blockIdx.x * blockDim.x + threadIdx.x;
    if (i < n) y[i] = (_Float16)x[i];
}

// ---------------------------------------------------------------- WMMA GEMM
// Y[M,N] = X[M,K] * W[N,K]^T + bias[N]   (all K-major). OutT = f16 or f32.
#define BM 128
#define BN 64
#define BK 32

template <typename OutT>
__global__ __launch_bounds__(256) void gemm_bias_wmma(
    const _Float16* __restrict__ X, const _Float16* __restrict__ W,
    const float* __restrict__ bias, OutT* __restrict__ Y,
    int M, int N, int K)
{
    __shared__ __align__(16) _Float16 As[BM][BK + 8];
    __shared__ __align__(16) _Float16 Bs[BN][BK + 8];

    const int tid  = threadIdx.x;
    const int lane = tid & 31;
    const int wave = tid >> 5;
    const int bm = blockIdx.x * BM;
    const int bn = blockIdx.y * BN;
    const int wr = (wave >> 1) * 32;   // 4 rows of waves
    const int wc = (wave & 1) * 32;    // 2 cols of waves

    v8f acc[2][2] = {};

    const int m16   = lane & 15;
    const int half  = lane >> 4;
    const int koffA = half * 16;   // A: one 16-half run
    const int koffB = half * 16;   // B: one 16-half run

    const int arow = tid >> 1, aseg = tid & 1;   // A: 2 thr/row, 16 halfs each
    const int brow = tid >> 2, bseg = tid & 3;   // B: 4 thr/row, 8 halfs each

    for (int k0 = 0; k0 < K; k0 += BK) {
        __syncthreads();
        async_copy_b128(X + (size_t)(bm + arow) * K + k0 + aseg * 16,      &As[arow][aseg * 16]);
        async_copy_b128(X + (size_t)(bm + arow) * K + k0 + aseg * 16 + 8,  &As[arow][aseg * 16 + 8]);
        async_copy_b128(W + (size_t)(bn + brow) * K + k0 + bseg * 8,       &Bs[brow][bseg * 8]);
        wait_async();
        __syncthreads();

        v16h a[2], b[2];
        #pragma unroll
        for (int si = 0; si < 2; ++si) {
            const _Float16* ap = &As[wr + si * 16 + m16][koffA];
            #pragma unroll
            for (int i = 0; i < 8; ++i) { a[si][i] = ap[i]; a[si][8 + i] = ap[i + 8]; }
        }
        #pragma unroll
        for (int sj = 0; sj < 2; ++sj) {
            const _Float16* bp = &Bs[wc + sj * 16 + m16][koffB];
            #pragma unroll
            for (int i = 0; i < 16; ++i) b[sj][i] = bp[i];
        }
        #pragma unroll
        for (int si = 0; si < 2; ++si)
            #pragma unroll
            for (int sj = 0; sj < 2; ++sj)
                acc[si][sj] = __builtin_amdgcn_wmma_f32_16x16x32_f16(
                    false, a[si], false, b[sj], (short)0, acc[si][sj], false, false);
    }

    // D layout: lane l, elem j -> row = j + 8*(l>>4), col = l&15
    const int rbase = bm + wr + 8 * half;
    const int cbase = bn + wc + m16;
    #pragma unroll
    for (int si = 0; si < 2; ++si) {
        #pragma unroll
        for (int sj = 0; sj < 2; ++sj) {
            int col = cbase + sj * 16;
            float bb = bias[col];
            #pragma unroll
            for (int j = 0; j < 8; ++j) {
                int row = rbase + si * 16 + j;
                Y[(size_t)row * N + col] = (OutT)(acc[si][sj][j] + bb);
            }
        }
    }
}

// ---------------------------------------------------------------- attention
// Single-pass online softmax, reference semantics:
//   m = running max of UNMASKED logits; p = exp(e-m)*maskbit; O /= (l + eps)
#define QT 64
#define KT 32

__global__ __launch_bounds__(256) void attn_wmma(
    const _Float16* __restrict__ Qp, const _Float16* __restrict__ Kp,
    const _Float16* __restrict__ Vp, const int* __restrict__ mask,
    _Float16* __restrict__ Ah)
{
    __shared__ __align__(16) _Float16 Qs[QT][HD + 8];
    __shared__ __align__(16) _Float16 Ks[KT][HD + 8];
    __shared__ __align__(16) _Float16 Vst[HD][KT + 8];   // V transposed: [d][key]
    __shared__ __align__(16) _Float16 Ps[QT][KT + 8];
    __shared__ float Mrow[QT], Lrow[QT], Scl[QT];
    __shared__ float Mpart[2][QT], Spart[2][QT];

    const int tid  = threadIdx.x;
    const int lane = tid & 31;
    const int wave = tid >> 5;
    const int m16  = lane & 15;
    const int half = lane >> 4;

    const int bh = blockIdx.y;           // 0..31
    const int b  = bh >> 4;
    const int h  = bh & 15;
    const int q0 = blockIdx.x * QT;

    const size_t qbase = ((size_t)b * NQ + q0) * DIMV + h * HD;
    const size_t kbase = (size_t)b * NK * DIMV + h * HD;

    {   // Q tile: 64 rows x 64 halfs; 4 threads/row, 16 halfs each (2x b128)
        int row = tid >> 2, seg = tid & 3;
        const _Float16* s = Qp + qbase + (size_t)row * DIMV + seg * 16;
        async_copy_b128(s,     &Qs[row][seg * 16]);
        async_copy_b128(s + 8, &Qs[row][seg * 16 + 8]);
    }
    if (tid < QT) { Mrow[tid] = -INFINITY; Lrow[tid] = 0.0f; }

    const int er = wave >> 1;            // E subtile row (0..3)
    const int ec = wave & 1;             // E subtile col (0..1)
    const int pr = wave >> 1;            // PV subtile row
    const int c0 = (wave & 1) * 2;       // PV col subtile base (0 or 2)

    v8f o[2] = {};
    const float inv_sqrt_dimV = 0.03125f;   // 1/sqrt(1024)

    const int kvrow = tid >> 3, kvseg = tid & 7;   // K/V: 8 thr/row, 8 halfs

    for (int kb = 0; kb < NK / KT; ++kb) {
        __syncthreads();
        {   // K block: async direct to LDS. V block: register-staged,
            // stored TRANSPOSED so P@V B-operand reads are contiguous.
            size_t src = kbase + (size_t)(kb * KT + kvrow) * DIMV + kvseg * 8;
            async_copy_b128(Kp + src, &Ks[kvrow][kvseg * 8]);
            union { uint4 u; _Float16 hh[8]; } uv;
            uv.u = *(const uint4*)(Vp + src);
            #pragma unroll
            for (int i = 0; i < 8; ++i) Vst[kvseg * 8 + i][kvrow] = uv.hh[i];
        }
        wait_async();
        __syncthreads();

        // ---- E subtile (16x16) per wave, K = head_dim 64 = 2 WMMAs
        v8f e = {};
        #pragma unroll
        for (int dk = 0; dk < HD; dk += 32) {
            v16h a, bb;
            const _Float16* ap = &Qs[er * 16 + m16][dk + half * 16];
            #pragma unroll
            for (int i = 0; i < 8; ++i) { a[i] = ap[i]; a[8 + i] = ap[i + 8]; }
            const _Float16* bp = &Ks[ec * 16 + m16][dk + half * 16];
            #pragma unroll
            for (int i = 0; i < 16; ++i) bb[i] = bp[i];
            e = __builtin_amdgcn_wmma_f32_16x16x32_f16(false, a, false, bb,
                                                       (short)0, e, false, false);
        }

        // ---- block row-max (16 lanes hold the 16 columns of a row)
        float esc[8], rm[8];
        #pragma unroll
        for (int j = 0; j < 8; ++j) {
            esc[j] = e[j] * inv_sqrt_dimV;
            float v = esc[j];
            #pragma unroll
            for (int off = 1; off < 16; off <<= 1)
                v = fmaxf(v, __shfl_xor(v, off, 32));
            rm[j] = v;
        }
        if (m16 == 0) {
            #pragma unroll
            for (int j = 0; j < 8; ++j)
                Mpart[ec][er * 16 + j + 8 * half] = rm[j];
        }
        __syncthreads();
        if (tid < QT) {
            float mo = Mrow[tid];
            float mn = fmaxf(mo, fmaxf(Mpart[0][tid], Mpart[1][tid]));
            Mrow[tid] = mn;
            Scl[tid]  = __expf(mo - mn);   // 0 on first block (mo = -inf)
        }
        __syncthreads();

        // ---- P = exp(e - m)*maskbit, stage to LDS, partial row sums
        float rs[8];
        #pragma unroll
        for (int j = 0; j < 8; ++j) {
            int row  = er * 16 + j + 8 * half;
            int kcol = kb * KT + ec * 16 + m16;
            int mv   = mask[((size_t)b * NQ + q0 + row) * NK + kcol];
            float p  = mv ? __expf(esc[j] - Mrow[row]) : 0.0f;
            Ps[row][ec * 16 + m16] = (_Float16)p;
            float v = p;
            #pragma unroll
            for (int off = 1; off < 16; off <<= 1)
                v += __shfl_xor(v, off, 32);
            rs[j] = v;
        }
        if (m16 == 0) {
            #pragma unroll
            for (int j = 0; j < 8; ++j)
                Spart[ec][er * 16 + j + 8 * half] = rs[j];
        }
        // ---- rescale O accumulators (rows owned by this wave)
        #pragma unroll
        for (int j = 0; j < 8; ++j) {
            float s = Scl[pr * 16 + j + 8 * half];
            o[0][j] *= s; o[1][j] *= s;
        }
        __syncthreads();
        if (tid < QT)
            Lrow[tid] = Lrow[tid] * Scl[tid] + Spart[0][tid] + Spart[1][tid];

        // ---- O += P @ V   (K = 32 keys; 2 col-subtiles per wave)
        v16h pa;
        {
            const _Float16* ap = &Ps[pr * 16 + m16][half * 16];
            #pragma unroll
            for (int i = 0; i < 8; ++i) { pa[i] = ap[i]; pa[8 + i] = ap[i + 8]; }
        }
        #pragma unroll
        for (int t = 0; t < 2; ++t) {
            int c = c0 + t;
            v16h vb;
            const _Float16* vp_ = &Vst[c * 16 + m16][half * 16];
            #pragma unroll
            for (int i = 0; i < 16; ++i) vb[i] = vp_[i];
            o[t] = __builtin_amdgcn_wmma_f32_16x16x32_f16(false, pa, false, vb,
                                                          (short)0, o[t], false, false);
        }
    }
    __syncthreads();

    // ---- normalize by (l + eps) and write head slice (concat layout)
    #pragma unroll
    for (int t = 0; t < 2; ++t) {
        int col = (c0 + t) * 16 + m16;
        #pragma unroll
        for (int j = 0; j < 8; ++j) {
            int row = pr * 16 + j + 8 * half;
            float denom = Lrow[row] + 1e-8f;
            Ah[qbase + (size_t)row * DIMV + col] = (_Float16)(o[t][j] / denom);
        }
    }
}

// ---------------------------------------------------------------- launcher
extern "C" void kernel_launch(void* const* d_in, const int* in_sizes, int n_in,
                              void* d_out, int out_size, void* d_ws, size_t ws_size,
                              hipStream_t stream) {
    const float* Q    = (const float*)d_in[0];
    const float* K    = (const float*)d_in[1];
    const int*   mask = (const int*)  d_in[2];
    const float* Wq   = (const float*)d_in[3];
    const float* bq   = (const float*)d_in[4];
    const float* Wk   = (const float*)d_in[5];
    const float* bk   = (const float*)d_in[6];
    const float* Wv   = (const float*)d_in[7];
    const float* bv   = (const float*)d_in[8];
    const float* Wo   = (const float*)d_in[9];
    const float* bo   = (const float*)d_in[10];

    const int M  = BATCH * NQ;       // 4096
    const int NM = M * DIMV;         // 4M
    const int NW = DIMV * DIMV;      // 1M

    _Float16* p   = (_Float16*)d_ws;
    _Float16* Qh  = p;  p += NM;
    _Float16* Kh  = p;  p += NM;
    _Float16* Wqh = p;  p += NW;
    _Float16* Wkh = p;  p += NW;
    _Float16* Wvh = p;  p += NW;
    _Float16* Woh = p;  p += NW;
    _Float16* Qp  = p;  p += NM;
    _Float16* Kp  = p;  p += NM;
    _Float16* Vp  = p;  p += NM;
    _Float16* Ahh = p;  p += NM;

    const int T = 256;
    cvt_f32_f16<<<(NM + T - 1) / T, T, 0, stream>>>(Q,  Qh,  NM);
    cvt_f32_f16<<<(NM + T - 1) / T, T, 0, stream>>>(K,  Kh,  NM);
    cvt_f32_f16<<<(NW + T - 1) / T, T, 0, stream>>>(Wq, Wqh, NW);
    cvt_f32_f16<<<(NW + T - 1) / T, T, 0, stream>>>(Wk, Wkh, NW);
    cvt_f32_f16<<<(NW + T - 1) / T, T, 0, stream>>>(Wv, Wvh, NW);
    cvt_f32_f16<<<(NW + T - 1) / T, T, 0, stream>>>(Wo, Woh, NW);

    dim3 gg(M / BM, DIMV / BN);
    gemm_bias_wmma<_Float16><<<gg, 256, 0, stream>>>(Qh, Wqh, bq, Qp, M, DIMV, DIMV);
    gemm_bias_wmma<_Float16><<<gg, 256, 0, stream>>>(Kh, Wkh, bk, Kp, M, DIMV, DIMV);
    gemm_bias_wmma<_Float16><<<gg, 256, 0, stream>>>(Kh, Wvh, bv, Vp, M, DIMV, DIMV);

    attn_wmma<<<dim3(NQ / QT, BATCH * HEADS), 256, 0, stream>>>(Qp, Kp, Vp, mask, Ahh);

    gemm_bias_wmma<float><<<gg, 256, 0, stream>>>(Ahh, Woh, bo, (float*)d_out,
                                                  M, DIMV, DIMV);
}
